// NetVLADPredictor_84258668413365
// MI455X (gfx1250) — matrix operand order repaired and checked by reading, not compiled, not run on hardware
//
#include <hip/hip_runtime.h>
#include <hip/hip_bf16.h>

typedef __attribute__((ext_vector_type(16))) _Float16 v16h;
typedef __attribute__((ext_vector_type(8)))  _Float16 v8h;
typedef __attribute__((ext_vector_type(8)))  float    v8f;

#define B_ITEMS 4096
#define NPTS    64
#define DIM     32
#define KCLUST  64
#define GD      2048      // KCLUST*DIM
#define NROW    4096
#define TOPK    11
#define ALPHA_C 100.0f
#define BETA_C  0.5f

// ---------------- Stage 1: NetVLAD (one block per item) ----------------
__global__ __launch_bounds__(256)
void netvlad_kernel(const float* __restrict__ desc,
                    const float* __restrict__ cent,
                    float* __restrict__ g,
                    _Float16* __restrict__ gh) {
  const int b = blockIdx.x;
  const int tid = threadIdx.x;
  __shared__ float sx[NPTS * DIM];     // normalized local descriptors
  __shared__ float sc[KCLUST * DIM];   // centroids
  __shared__ float acn[KCLUST];        // alpha * ||c_k||
  __shared__ float lg[KCLUST * NPTS];  // logits -> soft-assign
  __shared__ float asum[KCLUST];
  __shared__ float sv[KCLUST * DIM];   // vlad
  __shared__ float red[256];

  for (int p = tid; p < NPTS * DIM; p += 256) sx[p] = desc[(size_t)b * NPTS * DIM + p];
  for (int p = tid; p < KCLUST * DIM; p += 256) sc[p] = cent[p];
  __syncthreads();

  if (tid < NPTS) {                         // normalize_input rows
    float s = 0.f;
    for (int d = 0; d < DIM; ++d) { float v = sx[tid*DIM+d]; s += v*v; }
    float inv = 1.0f / fmaxf(sqrtf(s), 1e-12f);
    for (int d = 0; d < DIM; ++d) sx[tid*DIM+d] *= inv;
  }
  if (tid >= 64 && tid < 64 + KCLUST) {     // alpha*||c_k||
    int k = tid - 64;
    float s = 0.f;
    for (int d = 0; d < DIM; ++d) { float v = sc[k*DIM+d]; s += v*v; }
    acn[k] = ALPHA_C * sqrtf(s);
  }
  __syncthreads();

  for (int p = tid; p < KCLUST * NPTS; p += 256) {   // logits[k][n]
    int k = p >> 6, n = p & 63;
    float dot = 0.f;
    for (int d = 0; d < DIM; ++d) dot += sx[n*DIM+d] * sc[k*DIM+d];
    lg[k*NPTS + n] = 2.0f * ALPHA_C * dot - acn[k];
  }
  __syncthreads();

  if (tid < NPTS) {                         // softmax over k for column n
    int n = tid;
    float m = -3.4e38f;
    for (int k = 0; k < KCLUST; ++k) m = fmaxf(m, lg[k*NPTS+n]);
    float s = 0.f;
    for (int k = 0; k < KCLUST; ++k) { float e = expf(lg[k*NPTS+n]-m); lg[k*NPTS+n] = e; s += e; }
    float inv = 1.0f / s;
    for (int k = 0; k < KCLUST; ++k) lg[k*NPTS+n] *= inv;
  }
  __syncthreads();

  if (tid < KCLUST) {                       // sum_n a[k][n]
    float s = 0.f;
    for (int n = 0; n < NPTS; ++n) s += lg[tid*NPTS+n];
    asum[tid] = s;
  }
  __syncthreads();

  for (int p = tid; p < KCLUST * DIM; p += 256) {    // vlad[k][d]
    int k = p >> 5, d = p & 31;
    float acc = 0.f;
    for (int n = 0; n < NPTS; ++n) acc += lg[k*NPTS+n] * sx[n*DIM+d];
    sv[p] = acc - asum[k] * sc[k*DIM+d];
  }
  __syncthreads();

  if (tid < KCLUST) {                       // intra-normalize rows
    float s = 0.f;
    for (int d = 0; d < DIM; ++d) { float v = sv[tid*DIM+d]; s += v*v; }
    float inv = 1.0f / fmaxf(sqrtf(s), 1e-12f);
    for (int d = 0; d < DIM; ++d) sv[tid*DIM+d] *= inv;
  }
  __syncthreads();

  float ss = 0.f;                           // global L2 norm
  for (int p = tid; p < GD; p += 256) { float v = sv[p]; ss += v*v; }
  red[tid] = ss;
  __syncthreads();
  for (int s = 128; s > 0; s >>= 1) {
    if (tid < s) red[tid] += red[tid + s];
    __syncthreads();
  }
  float inv = 1.0f / fmaxf(sqrtf(red[0]), 1e-12f);
  for (int p = tid; p < GD; p += 256) {
    float v = sv[p] * inv;
    g[(size_t)b * GD + p]  = v;
    gh[(size_t)b * GD + p] = (_Float16)v;
  }
}

// ---------------- Stage 2/5: WMMA Gram (S = G * G^T), optional epilogue ----------------
// Each wave computes a 64x64 output tile as 4x4 WMMA tiles (16 wmma per K-step,
// 1:1 b128-load:wmma ratio). 4 waves/block -> 128x128 macro-tile, grid 32x32.
// mode 0: out[r][c] = dot(g_r, g_c)
// mode 1: out[r][c] = (r==c) ? 0 : (2 - sqrt(max(n_r + n_c - 2*dot, eps))) / 2
__global__ __launch_bounds__(128)
void gram_kernel(const _Float16* __restrict__ gh,
                 float* __restrict__ out,
                 const float* __restrict__ norms,
                 int mode) {
  const int lane = threadIdx.x & 31;
  const int wave = threadIdx.x >> 5;
  const int lr   = lane & 15;
  const int kh   = lane >> 4;          // K-half selector (contiguous 16-wide K halves used here)
  const int row0 = blockIdx.y * 128 + (wave >> 1) * 64;
  const int col0 = blockIdx.x * 128 + (wave & 1) * 64;

  v8f acc[4][4] = {};
  for (int kb = 0; kb < GD; kb += 32) {
    // B 32x16 fragments: column (col0+j*16+lr) of G^T == that row of G;
    // lane<16 holds contiguous K {kb..kb+15}, lane>=16 K {kb+16..kb+31}
    v16h bm[4];
#pragma unroll
    for (int j = 0; j < 4; ++j) {
      const _Float16* bp = gh + (size_t)(col0 + j*16 + lr) * GD + kb + kh*16;
      bm[j] = *(const v16h*)bp;
      __builtin_prefetch(bp + 32, 0, 3);   // next K-block -> global_prefetch_b8
    }
#pragma unroll
    for (int i = 0; i < 4; ++i) {
      // A 16x32 fragment: lane<16 rows hold contiguous K {kb..kb+15}, lane>=16 K {kb+16..kb+31}
      const _Float16* ap = gh + (size_t)(row0 + i*16 + lr) * GD + kb + kh*16;
      v8h lo = *(const v8h*)ap;
      v8h hi = *(const v8h*)(ap + 8);
      __builtin_prefetch(ap + 32, 0, 3);
      v16h a;
#pragma unroll
      for (int e = 0; e < 8; ++e) { a[e] = lo[e]; a[8+e] = hi[e]; }
#pragma unroll
      for (int j = 0; j < 4; ++j)
        acc[i][j] = __builtin_amdgcn_wmma_f32_16x16x32_f16(
            false, a, false, bm[j], (short)0, acc[i][j], false, false);
    }
  }

#pragma unroll
  for (int i = 0; i < 4; ++i)
#pragma unroll
    for (int j = 0; j < 4; ++j) {
      const int rbase = row0 + i*16 + kh*8;   // C/D layout: VGPR v -> M = v + 8*half
      const int c     = col0 + j*16 + lr;
#pragma unroll
      for (int v = 0; v < 8; ++v) {
        const int r = rbase + v;
        float dot = acc[i][j][v];
        float val;
        if (mode == 0) {
          val = dot;
        } else {
          float sq  = norms[r] + norms[c] - 2.0f * dot;
          float dis = sqrtf(fmaxf(sq, 1e-12f));
          val = (r == c) ? 0.0f : (2.0f - dis) * 0.5f;
        }
        out[(size_t)r * NROW + c] = val;
      }
    }
}

// ---------------- Stage 3: iterative top-11 per row ----------------
__global__ __launch_bounds__(256)
void topk_kernel(const float* __restrict__ S,
                 int* __restrict__ idx,
                 float* __restrict__ wv) {
  const int i = blockIdx.x;
  const int tid = threadIdx.x;
  __shared__ float row[NROW];
  __shared__ float rv[256];
  __shared__ int   ri[256];

  for (int j = tid; j < NROW; j += 256) row[j] = S[(size_t)i * NROW + j];
  __syncthreads();

  for (int t = 0; t < TOPK; ++t) {
    float bv = -3.4e38f; int bi = 0;
    for (int j = tid; j < NROW; j += 256)
      if (row[j] > bv) { bv = row[j]; bi = j; }
    rv[tid] = bv; ri[tid] = bi;
    __syncthreads();
    for (int s = 128; s > 0; s >>= 1) {
      if (tid < s && rv[tid + s] > rv[tid]) { rv[tid] = rv[tid + s]; ri[tid] = ri[tid + s]; }
      __syncthreads();
    }
    if (tid == 0) {
      idx[i * TOPK + t] = ri[0];
      wv[i * TOPK + t]  = rv[0] * (t == 0 ? 1.0f : BETA_C);  // w = g_i . neighbor, beta-scaled
      row[ri[0]] = -3.4e38f;
    }
    __syncthreads();
  }
}

// ---------------- Stage 4: neighbor-weighted refinement ----------------
__global__ __launch_bounds__(256)
void refine_kernel(const float* __restrict__ g,
                   const int* __restrict__ idx,
                   const float* __restrict__ wv,
                   _Float16* __restrict__ gh2,
                   float* __restrict__ norms) {
  const int i = blockIdx.x;
  const int tid = threadIdx.x;
  __shared__ int   sidx[TOPK];
  __shared__ float sw[TOPK];
  __shared__ float red[256];

  if (tid < TOPK) { sidx[tid] = idx[i * TOPK + tid]; sw[tid] = wv[i * TOPK + tid]; }
  __syncthreads();

  float wsum = 0.f;
  for (int j = 0; j < TOPK; ++j) wsum += sw[j];
  const float winv = 1.0f / wsum;

  float ss = 0.f;
  for (int d = tid; d < GD; d += 256) {
    float acc = 0.f;
    for (int j = 0; j < TOPK; ++j) acc += sw[j] * g[(size_t)sidx[j] * GD + d];
    float v = acc * winv;
    gh2[(size_t)i * GD + d] = (_Float16)v;
    ss += v * v;
  }
  red[tid] = ss;
  __syncthreads();
  for (int s = 128; s > 0; s >>= 1) {
    if (tid < s) red[tid] += red[tid + s];
    __syncthreads();
  }
  if (tid == 0) norms[i] = red[0];
}

// ---------------- Launch ----------------
extern "C" void kernel_launch(void* const* d_in, const int* in_sizes, int n_in,
                              void* d_out, int out_size, void* d_ws, size_t ws_size,
                              hipStream_t stream) {
  (void)in_sizes; (void)n_in; (void)out_size; (void)ws_size;
  const float* desc = (const float*)d_in[0];   // [4096, 64, 32]
  const float* cent = (const float*)d_in[1];   // [64, 32]
  float* out = (float*)d_out;                  // [4096, 4096]
  char* ws = (char*)d_ws;

  // Workspace layout (~48.5 MB):
  float*    G   = (float*)(ws);                          // 4096*2048 f32 = 32 MB
  _Float16* GH  = (_Float16*)(ws + 33554432);            // 4096*2048 f16 = 16 MB (reused for refined)
  int*      IDX = (int*)(ws + 50331648);                 // 4096*11 int
  float*    WV  = (float*)(ws + 50331648 + NROW*TOPK*4); // 4096*11 f32
  float*    NRM = (float*)(ws + 50331648 + NROW*TOPK*8); // 4096 f32

  // 1) NetVLAD -> g (f32) + gh (f16)
  netvlad_kernel<<<B_ITEMS, 256, 0, stream>>>(desc, cent, G, GH);

  // 2) Gram S = g.g^T via WMMA, stored in d_out as scratch (same size as output)
  gram_kernel<<<dim3(32, 32), 128, 0, stream>>>(GH, out, NRM, 0);

  // 3) top-11 neighbors + weights per row (largest similarity == smallest distance)
  topk_kernel<<<NROW, 256, 0, stream>>>(out, IDX, WV);

  // 4) weighted refinement -> refined f16 (reusing GH) + row norms
  refine_kernel<<<NROW, 256, 0, stream>>>(G, IDX, WV, GH, NRM);

  // 5) final overlap = (2 - cdist)/2 with zero diagonal, written over d_out
  gram_kernel<<<dim3(32, 32), 128, 0, stream>>>(GH, out, NRM, 1);
}
